// AdapterModel_43190191128808
// MI455X (gfx1250) — matrix operand order repaired and checked by reading, not compiled, not run on hardware
//
#include <hip/hip_runtime.h>

#define HH   32
#define DD   64
#define SEQD 3072
#define AAD  1280
#define CTXD 768
#define BB   8
#define NN   896
#define JJ   512
#define EE   2048   // H*D

typedef __attribute__((ext_vector_type(16))) __bf16 v16bf;
typedef __attribute__((ext_vector_type(2)))  __bf16 v2bf;
typedef __attribute__((ext_vector_type(8)))  float  v8f;

// Native bf16 converts: lets the backend use packed cvt instructions.
__device__ __forceinline__ unsigned int pk_bf16(float a, float b) {
  v2bf r;
  r[0] = (__bf16)a;
  r[1] = (__bf16)b;
  return __builtin_bit_cast(unsigned int, r);
}
__device__ __forceinline__ unsigned short bf1(float a) {
  __bf16 x = (__bf16)a;
  return __builtin_bit_cast(unsigned short, x);
}

// ---------------------------------------------------------------------------
// Elementwise f32 -> bf16 (4 elements / thread).
// ---------------------------------------------------------------------------
__global__ __launch_bounds__(256) void cvt_bf16_kernel(
    const float* __restrict__ in, unsigned short* __restrict__ out, int n4)
{
  const int i = blockIdx.x * 256 + threadIdx.x;
  if (i < n4) {
    const float4 v = ((const float4*)in)[i];
    uint2 o;
    o.x = pk_bf16(v.x, v.y);
    o.y = pk_bf16(v.z, v.w);
    ((uint2*)out)[i] = o;
  }
}

// ---------------------------------------------------------------------------
// W (K x E=2048) f32  ->  Wt (E x K) bf16, 32x32 LDS-tiled transpose.
// ---------------------------------------------------------------------------
__global__ __launch_bounds__(256) void transpose_cvt_kernel(
    const float* __restrict__ W, unsigned short* __restrict__ Wt, int K)
{
  __shared__ unsigned short tile[32][36];
  const int k0 = blockIdx.x * 32;
  const int e0 = blockIdx.y * 32;
  const int t  = threadIdx.x;
  {
    const int kr = t >> 3;
    const int e4 = (t & 7) * 4;
    const float4 v = *(const float4*)&W[(size_t)(k0 + kr) * EE + e0 + e4];
    tile[kr][e4 + 0] = bf1(v.x);
    tile[kr][e4 + 1] = bf1(v.y);
    tile[kr][e4 + 2] = bf1(v.z);
    tile[kr][e4 + 3] = bf1(v.w);
  }
  __syncthreads();
  {
    const int er = t >> 3;
    const int k4 = (t & 7) * 4;
    uint2 o;
    o.x = (unsigned int)tile[k4 + 0][er] | ((unsigned int)tile[k4 + 1][er] << 16);
    o.y = (unsigned int)tile[k4 + 2][er] | ((unsigned int)tile[k4 + 3][er] << 16);
    *(uint2*)&Wt[(size_t)(e0 + er) * K + k0 + k4] = o;
  }
}

// ---------------------------------------------------------------------------
// C = A(MxK) @ Wt(ExK)^T + bias, fused per-64-col (per-head) L2 norm.
// All-BF16 inputs.  Tile: 128(M) x 128(E), 8 waves each 32x64 (2x4 WMMA).
// ---------------------------------------------------------------------------
__global__ __launch_bounds__(256) void gemm_bf16_norm_kernel(
    const unsigned short* __restrict__ Abf,   // (B, M, K) bf16
    const unsigned short* __restrict__ Wt,    // (E, K)    bf16 (transposed)
    const float* __restrict__ bias,           // (E)
    unsigned short* __restrict__ outbf,       // (B, M, E) bf16, l2-normalized
    int M, int K)
{
  __shared__ unsigned short Asm[128 * 40];    // 128 x 32 bf16, pad 40
  __shared__ unsigned short Bsm[128 * 40];    // 128(n) x 32(k) bf16, pad 40
  __shared__ float          Csm[128 * 66];    // 128 x 64 half-tile f32, pad 66

  const int e0   = blockIdx.x * 128;
  const int m0   = blockIdx.y * 128;
  const int b    = blockIdx.z;
  const int t    = threadIdx.x;
  const int lane = t & 31;
  const int w    = t >> 5;
  const int wm   = (w >> 1) * 32;   // 0,32,64,96
  const int wn   = (w & 1) * 64;    // 0,64
  const int r    = lane & 15;
  const int hi   = lane >> 4;

  // staging assignment: each thread copies 16 bf16 of one row (2x uint4)
  const int srow = t >> 1;
  const int sc0  = (t & 1) * 16;
  const unsigned short* aSrc = Abf + (size_t)(b * M + m0 + srow) * K + sc0;
  const unsigned short* bSrc = Wt  + (size_t)(e0 + srow) * K + sc0;
  uint4* aDst = (uint4*)&Asm[srow * 40 + sc0];
  uint4* bDst = (uint4*)&Bsm[srow * 40 + sc0];

  v8f acc[2][4] = {};

  for (int k0 = 0; k0 < K; k0 += 32) {
    aDst[0] = *(const uint4*)(aSrc + k0);
    aDst[1] = *(const uint4*)(aSrc + k0 + 8);
    bDst[0] = *(const uint4*)(bSrc + k0);
    bDst[1] = *(const uint4*)(bSrc + k0 + 8);
    // branchless prefetch of next K-slice (clamped; emits global_prefetch)
    const int kn = (k0 + 32 < K) ? (k0 + 32) : k0;
    __builtin_prefetch(aSrc + kn, 0, 1);
    __builtin_prefetch(bSrc + kn, 0, 1);
    __syncthreads();

    v16bf afrag[2];
#pragma unroll
    for (int sm = 0; sm < 2; ++sm) {
      const unsigned short* ap = &Asm[(wm + sm * 16 + r) * 40];
      const int kb = hi * 8;
      ((uint4*)&afrag[sm])[0] = *(const uint4*)&ap[kb];       // K = kb..kb+7
      ((uint4*)&afrag[sm])[1] = *(const uint4*)&ap[kb + 16];  // K = kb+16..kb+23
    }
#pragma unroll
    for (int sn = 0; sn < 4; ++sn) {
      v16bf bfrag;
      const unsigned short* bp = &Bsm[(wn + sn * 16 + r) * 40 + hi * 8];
      ((uint4*)&bfrag)[0] = ((const uint4*)bp)[0];
      ((uint4*)&bfrag)[1] = ((const uint4*)bp)[2];
#pragma unroll
      for (int sm = 0; sm < 2; ++sm)
        acc[sm][sn] = __builtin_amdgcn_wmma_f32_16x16x32_bf16(
            false, afrag[sm], false, bfrag, (short)0, acc[sm][sn], false, false);
    }
    __syncthreads();
  }

  // ---- epilogue in two 64-column half-passes: bias + per-head L2 norm ----
  for (int p = 0; p < 2; ++p) {
    if ((w & 1) == p) {
#pragma unroll
      for (int sn = 0; sn < 4; ++sn) {
        const int colL = sn * 16 + r;
        const float bv = bias[e0 + p * 64 + colL];
#pragma unroll
        for (int sm = 0; sm < 2; ++sm)
#pragma unroll
          for (int i = 0; i < 8; ++i)
            Csm[(wm + sm * 16 + hi * 8 + i) * 66 + colL] = acc[sm][sn][i] + bv;
      }
    }
    __syncthreads();
    if (t < 128) {
      const float* rowp = &Csm[t * 66];
      float ss = 0.f;
#pragma unroll
      for (int c = 0; c < 64; ++c) ss += rowp[c] * rowp[c];
      const float scale = 1.f / fmaxf(sqrtf(ss), 1e-12f);
      unsigned int* dst =
          (unsigned int*)&outbf[(size_t)(b * M + m0 + t) * EE + e0 + p * 64];
#pragma unroll
      for (int c = 0; c < 32; ++c)
        dst[c] = pk_bf16(rowp[2 * c] * scale, rowp[2 * c + 1] * scale);
    }
    __syncthreads();
  }
}

// ---------------------------------------------------------------------------
// interactions[b,h] = seq_n(896x64) @ aa_n(512x64)^T, fused streaming
// logavgexp over J=512 (mask-aware).  Grid: (Ntile=7, H=32, B=8).
// ---------------------------------------------------------------------------
__global__ __launch_bounds__(256) void inter_kernel(
    const unsigned short* __restrict__ seqn,   // (B, N, E) bf16 normalized
    const unsigned short* __restrict__ aan,    // (B, J, E) bf16 normalized
    const unsigned char*  __restrict__ mask,   // (B, J) bool
    float* __restrict__ inter)                 // (B, N, H)
{
  __shared__ unsigned short Asm[128 * 72];
  __shared__ unsigned short Bsm[64 * 72];
  __shared__ float          Ssm[128 * 65];

  const int n0   = blockIdx.x * 128;
  const int h    = blockIdx.y;
  const int b    = blockIdx.z;
  const int t    = threadIdx.x;
  const int lane = t & 31;
  const int w    = t >> 5;
  const int wm   = (w >> 1) * 32;
  const int wn   = (w & 1) * 32;
  const int r    = lane & 15;
  const int hi   = lane >> 4;

  {
    const int row = t >> 1;
    const int c0  = (t & 1) * 32;
    const uint4* src = (const uint4*)&seqn[(size_t)(b * NN + n0 + row) * EE + h * 64 + c0];
    uint4* dst = (uint4*)&Asm[row * 72 + c0];
    dst[0] = src[0]; dst[1] = src[1]; dst[2] = src[2]; dst[3] = src[3];
  }
  __syncthreads();

  v16bf afrag[2][2];
#pragma unroll
  for (int sm = 0; sm < 2; ++sm) {
    const unsigned short* ap = &Asm[(wm + sm * 16 + r) * 72];
#pragma unroll
    for (int ks = 0; ks < 2; ++ks) {
      const int kb = ks * 32 + hi * 8;
      ((uint4*)&afrag[sm][ks])[0] = *(const uint4*)&ap[kb];
      ((uint4*)&afrag[sm][ks])[1] = *(const uint4*)&ap[kb + 16];
    }
  }

  float m_run = -3.4e38f;
  float s_run = 0.f;
  int   nj    = 0;

  for (int j0 = 0; j0 < JJ; j0 += 64) {
    __syncthreads();
    {
      const int row = t >> 2;
      const int c0  = (t & 3) * 16;
      const uint4* src = (const uint4*)&aan[(size_t)(b * JJ + j0 + row) * EE + h * 64 + c0];
      uint4* dst = (uint4*)&Bsm[row * 72 + c0];
      dst[0] = src[0]; dst[1] = src[1];
    }
    __syncthreads();

    v8f acc[2][2] = {};
#pragma unroll
    for (int sn = 0; sn < 2; ++sn) {
#pragma unroll
      for (int ks = 0; ks < 2; ++ks) {
        v16bf bfrag;
        const unsigned short* bp = &Bsm[(wn + sn * 16 + r) * 72 + ks * 32 + hi * 8];
        ((uint4*)&bfrag)[0] = ((const uint4*)bp)[0];
        ((uint4*)&bfrag)[1] = ((const uint4*)bp)[2];
#pragma unroll
        for (int sm = 0; sm < 2; ++sm)
          acc[sm][sn] = __builtin_amdgcn_wmma_f32_16x16x32_bf16(
              false, afrag[sm][ks], false, bfrag, (short)0, acc[sm][sn], false, false);
      }
    }
#pragma unroll
    for (int sm = 0; sm < 2; ++sm)
#pragma unroll
      for (int sn = 0; sn < 2; ++sn) {
        const int colL = wn + sn * 16 + r;
#pragma unroll
        for (int i = 0; i < 8; ++i)
          Ssm[(wm + sm * 16 + hi * 8 + i) * 65 + colL] = acc[sm][sn][i];
      }
    __syncthreads();

    if (t < 128) {
      const float* rp = &Ssm[t * 65];
      const unsigned char* mp = &mask[(size_t)b * JJ + j0];
      for (int c = 0; c < 64; ++c) {
        if (mp[c]) {
          const float v = rp[c] * 100.f;   // t = S / temp, temp = 0.01
          if (v > m_run) { s_run = s_run * expf(m_run - v) + 1.f; m_run = v; }
          else           { s_run += expf(v - m_run); }
          ++nj;
        }
      }
    }
  }

  if (t < 128) {
    const float fn  = (float)nj;
    const float sc  = fmaxf(s_run, 1e-20f);
    const float res = (logf(sc / fn + 1e-20f) + m_run - logf(fn)) * 0.01f;
    inter[(size_t)(b * NN + n0 + t) * HH + h] = res;
  }
}

// ---------------------------------------------------------------------------
// gating = sigmoid(ctx @ ctx_w + ctx_b); wv[d] = sum_e tlw[d,e]*g[d,e]*pred_w[e]
// pred[n] = softplus(inter[n,:] . wv + pred_b).  One block per batch.
// ---------------------------------------------------------------------------
__global__ __launch_bounds__(256) void final_kernel(
    const float* __restrict__ ctx,     // (B, 768)
    const float* __restrict__ ctx_w,   // (768, 1024)
    const float* __restrict__ ctx_b,   // (1024)
    const float* __restrict__ tlw,     // (32, 32)
    const float* __restrict__ pred_w,  // (32, 1)
    const float* __restrict__ pred_b,  // (1)
    const float* __restrict__ inter,   // (B, N, 32)
    float* __restrict__ out)           // (B, N)
{
  __shared__ float cx[CTXD];
  __shared__ float gg[HH * HH];
  __shared__ float wv[HH];

  const int b = blockIdx.x;
  const int t = threadIdx.x;

  for (int i = t; i < CTXD; i += 256) cx[i] = ctx[b * CTXD + i];
  __syncthreads();

  for (int o = t; o < HH * HH; o += 256) {
    float a = ctx_b[o];
    for (int k = 0; k < CTXD; ++k) a += cx[k] * ctx_w[(size_t)k * (HH * HH) + o];
    gg[o] = 1.f / (1.f + expf(-a));
  }
  __syncthreads();

  if (t < HH) {
    float a = 0.f;
#pragma unroll
    for (int e = 0; e < HH; ++e) a += tlw[t * HH + e] * gg[t * HH + e] * pred_w[e];
    wv[t] = a;
  }
  __syncthreads();

  const float pb = pred_b[0];
  for (int n = t; n < NN; n += 256) {
    const float* ip = &inter[(size_t)(b * NN + n) * HH];
    float a = pb;
#pragma unroll
    for (int d = 0; d < HH; ++d) a += ip[d] * wv[d];
    out[b * NN + n] = fmaxf(a, 0.f) + log1pf(expf(-fabsf(a)));   // softplus
  }
}

// ---------------------------------------------------------------------------
extern "C" void kernel_launch(void* const* d_in, const int* in_sizes, int n_in,
                              void* d_out, int out_size, void* d_ws, size_t ws_size,
                              hipStream_t stream) {
  const float*         seq_embed = (const float*)d_in[0];
  const float*         aa_embed  = (const float*)d_in[1];
  const float*         ctx       = (const float*)d_in[2];
  const unsigned char* aa_mask   = (const unsigned char*)d_in[3];
  const float*         seq_w     = (const float*)d_in[4];
  const float*         seq_b     = (const float*)d_in[5];
  const float*         aa_w      = (const float*)d_in[6];
  const float*         aa_b      = (const float*)d_in[7];
  const float*         tlw       = (const float*)d_in[8];
  const float*         ctx_w     = (const float*)d_in[9];
  const float*         ctx_b     = (const float*)d_in[10];
  const float*         pred_w    = (const float*)d_in[11];
  const float*         pred_b    = (const float*)d_in[12];

  // workspace layout (bytes)
  char* ws = (char*)d_ws;
  const size_t seqe_bytes = (size_t)BB * NN * SEQD * 2;  // 44.0 MB bf16 seq_embed
  const size_t aae_bytes  = (size_t)BB * JJ * AAD * 2;   // 10.5 MB bf16 aa_embed
  const size_t swt_bytes  = (size_t)EE * SEQD * 2;       // 12.6 MB bf16 seq_w^T
  const size_t awt_bytes  = (size_t)EE * AAD * 2;        //  5.2 MB bf16 aa_w^T
  const size_t seqn_bytes = (size_t)BB * NN * EE * 2;    // 29.4 MB bf16 seq latent

  unsigned short* seqe = (unsigned short*)ws;
  unsigned short* aae  = (unsigned short*)(ws + seqe_bytes);
  unsigned short* swt  = (unsigned short*)(ws + seqe_bytes + aae_bytes);
  unsigned short* awt  = (unsigned short*)(ws + seqe_bytes + aae_bytes + swt_bytes);
  unsigned short* seqn = (unsigned short*)(ws + seqe_bytes + aae_bytes + swt_bytes + awt_bytes);
  float* inter = (float*)(ws + seqe_bytes + aae_bytes + swt_bytes + awt_bytes + seqn_bytes);
  // aa latent aliases the seq_embed bf16 region: seqe is dead once gemm #1 has
  // run, and gemm #2 (which writes aan) is stream-ordered after it.
  unsigned short* aan = seqe;

  // 1) one-time bf16 conversions / weight transposes (pure bandwidth)
  cvt_bf16_kernel<<<(BB * NN * SEQD / 4 + 255) / 256, 256, 0, stream>>>(
      seq_embed, seqe, BB * NN * SEQD / 4);
  cvt_bf16_kernel<<<(BB * JJ * AAD / 4 + 255) / 256, 256, 0, stream>>>(
      aa_embed, aae, BB * JJ * AAD / 4);
  transpose_cvt_kernel<<<dim3(SEQD / 32, EE / 32), 256, 0, stream>>>(seq_w, swt, SEQD);
  transpose_cvt_kernel<<<dim3(AAD / 32, EE / 32), 256, 0, stream>>>(aa_w, awt, AAD);

  // 2) WMMA GEMMs with fused bias + per-head l2norm
  gemm_bf16_norm_kernel<<<dim3(EE / 128, NN / 128, BB), 256, 0, stream>>>(
      seqe, swt, seq_b, seqn, NN, SEQD);
  gemm_bf16_norm_kernel<<<dim3(EE / 128, JJ / 128, BB), 256, 0, stream>>>(
      aae, awt, aa_b, aan, JJ, AAD);

  // 3) WMMA interactions + fused streaming logavgexp
  inter_kernel<<<dim3(NN / 128, HH, BB), 256, 0, stream>>>(seqn, aan, aa_mask, inter);

  // 4) tiny head-mixing epilogue
  final_kernel<<<BB, 256, 0, stream>>>(
      ctx, ctx_w, ctx_b, tlw, pred_w, pred_b, inter, (float*)d_out);
}
